// HopfieldRetrieve_13271448945091
// MI455X (gfx1250) — compile-verified, matrix operand order checked by reading, not run on hardware
//
#include <hip/hip_runtime.h>
#include <hip/hip_bf16.h>

typedef __attribute__((ext_vector_type(16))) _Float16 v16h;
typedef __attribute__((ext_vector_type(8)))  float    v8f;
typedef __attribute__((ext_vector_type(4)))  unsigned int u32x4;
typedef __attribute__((ext_vector_type(8)))  unsigned int u32x8;

#define D      1024
#define NSEQ   2048
#define BATCH  4
#define HEADS  16
#define DH     64
#define NORM   0.125f            // (1024/16)^-0.5
#define LOG2E  1.4426950408889634f

static __device__ __forceinline__ v8f wmma_f16(v16h a, v16h b, v8f c) {
    // D = A(16x32 f16) x B(32x16 f16) + C(16x16 f32)
    return __builtin_amdgcn_wmma_f32_16x16x32_f16(
        /*neg_a=*/false, a, /*neg_b=*/false, b,
        /*c_mod=*/(short)0, c, /*reuse_a=*/false, /*reuse_b=*/false);
}

// ---------------------------------------------------------------------------
// TDM: async 2D tile load Global -> LDS (32 keys x 64 dh, f16, row stride D).
// Descriptor per CDNA5 ISA; VADDR2/3 = NULL (2D tile).
// Tracked by TENSORcnt.
// ---------------------------------------------------------------------------
static __device__ __forceinline__ void tdm_load_tile(_Float16* ldsDst,
                                                     const _Float16* gsrc) {
    const unsigned           ldsAddr = (unsigned)(uintptr_t)ldsDst; // addr[31:0] = LDS byte offset
    const unsigned long long ga      = (unsigned long long)(uintptr_t)gsrc;
    u32x4 g0;
    g0[0] = 1u;                                               // count=1 (valid user D#)
    g0[1] = ldsAddr;                                          // lds_addr (bytes)
    g0[2] = (unsigned)ga;                                     // global_addr[31:0]
    g0[3] = ((unsigned)(ga >> 32) & 0x01FFFFFFu) | (2u << 30); // global_addr[56:32] | type=2
    u32x8 g1;
    g1[0] = 1u << 16;              // workgroup_mask=0, data_size=1 (2 bytes)
    g1[1] = (unsigned)DH << 16;    // tensor_dim0[15:0] = 64
    g1[2] = 32u << 16;             // tensor_dim0[31:16]=0 | tensor_dim1[15:0] = 32
    g1[3] = (unsigned)DH << 16;    // tensor_dim1[31:16]=0 | tile_dim0 = 64
    g1[4] = 32u;                   // tile_dim1 = 32, tile_dim2 = 0
    g1[5] = (unsigned)D;           // tensor_dim0_stride[31:0] = 1024 elements
    g1[6] = 0u;                    // stride hi / tensor_dim1_stride lo
    g1[7] = 0u;                    // tensor_dim1_stride hi
    asm volatile("tensor_load_to_lds %0, %1" :: "s"(g0), "s"(g1) : "memory");
}

// ---------------------------------------------------------------------------
// Kernel 1: convert the three fp32 weight matrices to f16 in workspace.
// grid = (D*D/256, 3)
// ---------------------------------------------------------------------------
__global__ void cvt_w_kernel(const float* __restrict__ Wq,
                             const float* __restrict__ Wk,
                             const float* __restrict__ Wv,
                             _Float16* __restrict__ Wh) {
    const int i = blockIdx.x * 256 + threadIdx.x;
    const float* src = (blockIdx.y == 0) ? Wq : ((blockIdx.y == 1) ? Wk : Wv);
    Wh[(size_t)blockIdx.y * D * D + i] = (_Float16)src[i];
}

// ---------------------------------------------------------------------------
// Kernel 2: projection GEMM + ReLU, fp32 X * f16 W^T -> f16.
// Register-blocked: one wave computes a 16x64 output strip, so the A
// fragment (and its fp32->f16 conversion) amortizes over 4 WMMAs.
// Q/K stored row-major [b*n, d]; V stored TRANSPOSED per batch Vt[b][d][n].
// grid = (1024, 3), block = 256 (8 waves)
// ---------------------------------------------------------------------------
__global__ __launch_bounds__(256) void proj_kernel(
    const float* __restrict__ q, const float* __restrict__ k,
    const float* __restrict__ v, const _Float16* __restrict__ Whall,
    _Float16* __restrict__ Qp, _Float16* __restrict__ Kp,
    _Float16* __restrict__ Vt) {
    const int proj = blockIdx.y;
    const float*    X = (proj == 0) ? q : ((proj == 1) ? k : v);
    const _Float16* W = Whall + (size_t)proj * D * D;

    const int lane = threadIdx.x & 31;
    const int wave = threadIdx.x >> 5;
    const int gw   = blockIdx.x * 8 + wave;        // 0 .. 8191
    const int tn   = gw >> 4;                      // row tile (0..511)
    const int cb   = (gw & 15) * 64;               // col base (0..960)
    const int lm   = lane & 15, lh = lane >> 4;

    const float* xrow = X + (size_t)(tn * 16 + lm) * D;   // A: row per lane

    v8f acc0 = {}, acc1 = {}, acc2 = {}, acc3 = {};
    for (int k0 = 0; k0 < D; k0 += 32) {
        const int off = k0 + lh * 16;
        // A fragment: 16 fp32 -> 16 f16 per lane (shared by 4 WMMAs)
        v16h a;
        const float4* xp = (const float4*)(xrow + off);
#pragma unroll
        for (int i = 0; i < 4; ++i) {
            float4 f = xp[i];
            a[4 * i + 0] = (_Float16)f.x; a[4 * i + 1] = (_Float16)f.y;
            a[4 * i + 2] = (_Float16)f.z; a[4 * i + 3] = (_Float16)f.w;
        }
        // 4 B fragments: contiguous 16 halves per lane each
        const _Float16* wbase = W + (size_t)(cb + lm) * D + off;
        v16h b0 = *(const v16h*)(wbase + (size_t)(0 * 16) * D);
        v16h b1 = *(const v16h*)(wbase + (size_t)(1 * 16) * D);
        v16h b2 = *(const v16h*)(wbase + (size_t)(2 * 16) * D);
        v16h b3 = *(const v16h*)(wbase + (size_t)(3 * 16) * D);
        acc0 = wmma_f16(a, b0, acc0);
        acc1 = wmma_f16(a, b1, acc1);
        acc2 = wmma_f16(a, b2, acc2);
        acc3 = wmma_f16(a, b3, acc3);
    }
    // ReLU + f16 store (C layout: row = r + 8*(lane>=16), col tile = lane%16)
    _Float16* QK = (proj == 0) ? Qp : Kp;
#pragma unroll
    for (int t = 0; t < 4; ++t) {
        const v8f* at  = (t == 0) ? &acc0 : (t == 1) ? &acc1 : (t == 2) ? &acc2 : &acc3;
        const int  col = cb + t * 16 + lm;
#pragma unroll
        for (int r = 0; r < 8; ++r) {
            const float    val = fmaxf((*at)[r], 0.0f);
            const _Float16 hv  = (_Float16)val;
            const int      row = tn * 16 + r + 8 * lh;    // global [b*n] row
            if (proj != 2) {
                QK[(size_t)row * D + col] = hv;
            } else {                                      // V: transposed store
                const int bb = row >> 11;                 // row / NSEQ
                const int nn = row & (NSEQ - 1);
                Vt[((size_t)bb * D + col) * NSEQ + nn] = hv;
            }
        }
    }
}

// ---------------------------------------------------------------------------
// Kernel 3: flash attention. One wave owns a 16-row Q tile of one (b,h).
// K chunks are DMA'd into double-buffered LDS by the Tensor Data Mover
// (async, TENSORcnt); V is pre-transposed so P*V B-fragments are contiguous
// global loads; P goes through LDS for the C-layout -> A-layout transpose.
// grid = B*H*(N/16)/8 = 1024, block = 256 (8 waves)
// ---------------------------------------------------------------------------
__global__ __launch_bounds__(256) void attn_kernel(
    const _Float16* __restrict__ Qp, const _Float16* __restrict__ Kp,
    const _Float16* __restrict__ Vt, float* __restrict__ out) {
    // per wave: 2 x (32x64) K buffers + 16x32 P tile = 4608 halves (9 KB)
    __shared__ __align__(32) _Float16 smem[8 * 4608];   // 72 KB / block
    const int lane = threadIdx.x & 31;
    const int wave = threadIdx.x >> 5;
    _Float16* ldsK = smem + wave * 4608;        // two 2048-half K buffers
    _Float16* ldsP = smem + wave * 4608 + 4096; // 16(row) x 32(col) P

    const int gw  = blockIdx.x * 8 + wave;      // 0 .. 8191
    const int b   = gw >> 11;                   // /(H * N/16)
    const int rem = gw & 2047;
    const int h   = rem >> 7;
    const int qt  = rem & 127;
    const int qb  = qt * 16;
    const int lm  = lane & 15, lh = lane >> 4;

    // Q A-fragments for the two K=32 chunks of dh=64
    const _Float16* qrow = Qp + ((size_t)(b * NSEQ + qb + lm)) * D + h * DH;
    v16h qa0 = *(const v16h*)(qrow + 0  + lh * 16);
    v16h qa1 = *(const v16h*)(qrow + 32 + lh * 16);

    // V^T fragment base: row (dh) = h*DH + t*16 + lm, contiguous along keys
    const _Float16* vtb = Vt + ((size_t)b * D + h * DH + lm) * NSEQ + lh * 16;

    const _Float16* kh = Kp + (size_t)b * NSEQ * D + h * DH;

    v8f O0 = {}, O1 = {}, O2 = {}, O3 = {};
    v8f mrow, lrow;
#pragma unroll
    for (int r = 0; r < 8; ++r) { mrow[r] = -1e30f; lrow[r] = 0.0f; }

    // prologue: DMA first K chunk into buffer 0
    tdm_load_tile(ldsK, kh);

    for (int j = 0; j < NSEQ; j += 32) {
        const int cur = (j >> 5) & 1;
        _Float16* kbuf = ldsK + cur * 2048;

        // current chunk's DMA complete
        __builtin_amdgcn_s_wait_tensorcnt(0);
        // kick off DMA of the next chunk into the other buffer
        if (j + 32 < NSEQ) {
            tdm_load_tile(ldsK + (1 - cur) * 2048, kh + (size_t)(j + 32) * D);
            __builtin_prefetch(vtb + j + 32, 0, 3);
        }

        // --- S = Q K^T ; K fragments from LDS (row-major 32x64) ---
        v16h kb00 = *(const v16h*)(kbuf + lm * 64       + lh * 16);
        v16h kb01 = *(const v16h*)(kbuf + lm * 64 + 32  + lh * 16);
        v16h kb10 = *(const v16h*)(kbuf + (16 + lm) * 64      + lh * 16);
        v16h kb11 = *(const v16h*)(kbuf + (16 + lm) * 64 + 32 + lh * 16);
        v8f S0 = {}, S1 = {};
        S0 = wmma_f16(qa0, kb00, S0); S0 = wmma_f16(qa1, kb01, S0);
        S1 = wmma_f16(qa0, kb10, S1); S1 = wmma_f16(qa1, kb11, S1);

        // --- online softmax: row stats via shfl_xor within 16-lane halves ---
#pragma unroll
        for (int r = 0; r < 8; ++r) { S0[r] *= NORM; S1[r] *= NORM; }
        v8f mnew, alpha;
#pragma unroll
        for (int r = 0; r < 8; ++r) {
            float m = fmaxf(S0[r], S1[r]);
            m = fmaxf(m, __shfl_xor(m, 1, 32));
            m = fmaxf(m, __shfl_xor(m, 2, 32));
            m = fmaxf(m, __shfl_xor(m, 4, 32));
            m = fmaxf(m, __shfl_xor(m, 8, 32));
            mnew[r]  = fmaxf(mrow[r], m);
            alpha[r] = exp2f((mrow[r] - mnew[r]) * LOG2E);
        }
#pragma unroll
        for (int r = 0; r < 8; ++r) {
            float p0 = exp2f((S0[r] - mnew[r]) * LOG2E);
            float p1 = exp2f((S1[r] - mnew[r]) * LOG2E);
            S0[r] = p0; S1[r] = p1;
            float s = p0 + p1;
            s += __shfl_xor(s, 1, 32);
            s += __shfl_xor(s, 2, 32);
            s += __shfl_xor(s, 4, 32);
            s += __shfl_xor(s, 8, 32);
            lrow[r] = alpha[r] * lrow[r] + s;
            mrow[r] = mnew[r];
        }
#pragma unroll
        for (int r = 0; r < 8; ++r) {
            O0[r] *= alpha[r]; O1[r] *= alpha[r];
            O2[r] *= alpha[r]; O3[r] *= alpha[r];
        }

        // --- P: C layout -> A layout through LDS ---
#pragma unroll
        for (int r = 0; r < 8; ++r) {
            const int row = r + 8 * lh;
            ldsP[row * 32 + lm]      = (_Float16)S0[r];
            ldsP[row * 32 + 16 + lm] = (_Float16)S1[r];
        }
        v16h pa = *(const v16h*)(ldsP + lm * 32 + lh * 16);

        // --- O += P V : V^T rows are contiguous along keys ---
        v16h vb0 = *(const v16h*)(vtb + (size_t)(0 * 16) * NSEQ + j);
        v16h vb1 = *(const v16h*)(vtb + (size_t)(1 * 16) * NSEQ + j);
        v16h vb2 = *(const v16h*)(vtb + (size_t)(2 * 16) * NSEQ + j);
        v16h vb3 = *(const v16h*)(vtb + (size_t)(3 * 16) * NSEQ + j);
        O0 = wmma_f16(pa, vb0, O0);
        O1 = wmma_f16(pa, vb1, O1);
        O2 = wmma_f16(pa, vb2, O2);
        O3 = wmma_f16(pa, vb3, O3);
    }

    // --- normalize and store fp32 output ---
    v8f invl;
#pragma unroll
    for (int r = 0; r < 8; ++r) invl[r] = 1.0f / lrow[r];
#pragma unroll
    for (int r = 0; r < 8; ++r) {
        const int    row  = qb + r + 8 * lh;
        float*       orow = out + ((size_t)(b * NSEQ + row)) * D + h * DH + lm;
        orow[0]  = O0[r] * invl[r];
        orow[16] = O1[r] * invl[r];
        orow[32] = O2[r] * invl[r];
        orow[48] = O3[r] * invl[r];
    }
}

// ---------------------------------------------------------------------------
extern "C" void kernel_launch(void* const* d_in, const int* in_sizes, int n_in,
                              void* d_out, int out_size, void* d_ws, size_t ws_size,
                              hipStream_t stream) {
    const float* q  = (const float*)d_in[0];
    const float* k  = (const float*)d_in[1];
    const float* v  = (const float*)d_in[2];
    const float* Wq = (const float*)d_in[3];
    const float* Wk = (const float*)d_in[4];
    const float* Wv = (const float*)d_in[5];
    float* out = (float*)d_out;

    _Float16* ws = (_Float16*)d_ws;
    const size_t offW = 0;                               // 3*D*D halves  (6 MB)
    const size_t offQ = offW + (size_t)3 * D * D;        // 16 MB
    const size_t offK = offQ + (size_t)BATCH * NSEQ * D; // 16 MB
    const size_t offV = offK + (size_t)BATCH * NSEQ * D; // 16 MB (transposed)

    // 1) weights fp32 -> f16
    cvt_w_kernel<<<dim3(D * D / 256, 3), 256, 0, stream>>>(Wq, Wk, Wv, ws + offW);
    // 2) projections + ReLU -> f16 Qp/Kp + transposed Vt
    proj_kernel<<<dim3(1024, 3), 256, 0, stream>>>(q, k, v, ws + offW,
                                                   ws + offQ, ws + offK, ws + offV);
    // 3) flash attention (TDM double-buffered K)
    attn_kernel<<<dim3(1024), 256, 0, stream>>>(ws + offQ, ws + offK, ws + offV, out);
}
